// KGINModel_77489799955012
// MI455X (gfx1250) — matrix operand order repaired and checked by reading, not compiled, not run on hardware
//
#include <hip/hip_runtime.h>
#include <hip/hip_bf16.h>
#include <math.h>

#define N_USERS   100000
#define N_ENT     200000
#define C_DIM     64
#define F_DIM     4
#define N_RELM1   9        // N_REL - 1
#define E_EDGES   1000000
#define NNZ_UI    1000000
#define EPS_F     1e-12f

typedef __attribute__((ext_vector_type(16))) _Float16 v16h;
typedef __attribute__((ext_vector_type(8)))  float    v8f;
typedef __attribute__((ext_vector_type(2)))  float    v2f;

// ---------------------------------------------------------------------------
// init: copy src -> cur and src -> res (residual accumulator in d_out)
// ---------------------------------------------------------------------------
__global__ void init_copy_kernel(const float* __restrict__ src,
                                 float* __restrict__ cur,
                                 float* __restrict__ res, int n) {
    for (int i = blockIdx.x * blockDim.x + threadIdx.x; i < n;
         i += gridDim.x * blockDim.x) {
        float v = src[i];
        cur[i] = v;
        res[i] = v;
    }
}

// ---------------------------------------------------------------------------
// cor = sum_{i<j} cos^2(att_i, att_j), att is [4 x 9]. Tiny -> 1 thread.
// ---------------------------------------------------------------------------
__global__ void cor_kernel(const float* __restrict__ att, float* __restrict__ out) {
    if (threadIdx.x == 0 && blockIdx.x == 0) {
        float n[F_DIM][N_RELM1];
        for (int f = 0; f < F_DIM; ++f) {
            float ss = 0.0f;
            for (int r = 0; r < N_RELM1; ++r) {
                float v = att[f * N_RELM1 + r];
                ss += v * v;
            }
            float inv = 1.0f / fmaxf(sqrtf(ss), EPS_F);
            for (int r = 0; r < N_RELM1; ++r)
                n[f][r] = att[f * N_RELM1 + r] * inv;
        }
        float s = 0.0f;
        for (int i = 0; i < F_DIM; ++i)
            for (int j = i + 1; j < F_DIM; ++j) {
                float d = 0.0f;
                for (int r = 0; r < N_RELM1; ++r) d += n[i][r] * n[j][r];
                s += d * d;
            }
        *out = s;
    }
}

// ---------------------------------------------------------------------------
// disen = softmax(att, axis=1) @ weight   [4 x 64]; 64 threads, 1 block
// ---------------------------------------------------------------------------
__global__ void disen_kernel(const float* __restrict__ att,
                             const float* __restrict__ wgt,
                             float* __restrict__ disen) {
    int t = threadIdx.x;   // 0..63 -> output column
    for (int f = 0; f < F_DIM; ++f) {
        float mx = -1e30f;
        for (int r = 0; r < N_RELM1; ++r) mx = fmaxf(mx, att[f * N_RELM1 + r]);
        float ex[N_RELM1];
        float sum = 0.0f;
        for (int r = 0; r < N_RELM1; ++r) {
            ex[r] = expf(att[f * N_RELM1 + r] - mx);
            sum += ex[r];
        }
        float inv = 1.0f / sum;
        float acc = 0.0f;
        for (int r = 0; r < N_RELM1; ++r)
            acc += (ex[r] * inv) * wgt[r * C_DIM + t];
        disen[f * C_DIM + t] = acc;
    }
}

// ---------------------------------------------------------------------------
// KG edge aggregation: agg[head] += entity_cur[tail] * weight[etype-1];
// cnt[head] += 1. 16 threads/edge, float4 lanes, f32 atomics (L2-resident).
// ---------------------------------------------------------------------------
__global__ void edge_agg_kernel(const int* __restrict__ head,
                                const int* __restrict__ tail,
                                const int* __restrict__ etype,
                                const float* __restrict__ ecur,
                                const float* __restrict__ wgt,
                                float* __restrict__ agg,
                                float* __restrict__ cnt, int E) {
    int t = blockIdx.x * blockDim.x + threadIdx.x;
    int e = t >> 4, c4 = t & 15;
    if (e >= E) return;
    int h  = head[e];
    int tl = tail[e];
    int r  = etype[e] - 1;
    float4 v = *((const float4*)(ecur + tl * C_DIM + c4 * 4));
    float4 w = *((const float4*)(wgt  + r  * C_DIM + c4 * 4));
    float* dst = agg + h * C_DIM + c4 * 4;
    atomicAdd(dst + 0, v.x * w.x);
    atomicAdd(dst + 1, v.y * w.y);
    atomicAdd(dst + 2, v.z * w.z);
    atomicAdd(dst + 3, v.w * w.w);
    if (c4 == 0) atomicAdd(cnt + h, 1.0f);
}

// ---------------------------------------------------------------------------
// user-item sparse matmul: agg_u[row] += entity_cur[col] * val
// ---------------------------------------------------------------------------
__global__ void ui_agg_kernel(const int* __restrict__ rows,
                              const int* __restrict__ cols,
                              const float* __restrict__ vals,
                              const float* __restrict__ ecur,
                              float* __restrict__ aggu, int nnz) {
    int t = blockIdx.x * blockDim.x + threadIdx.x;
    int i = t >> 4, c4 = t & 15;
    if (i >= nnz) return;
    int r = rows[i];
    int c = cols[i];
    float vv = vals[i];
    float4 e = *((const float4*)(ecur + c * C_DIM + c4 * 4));
    float* dst = aggu + r * C_DIM + c4 * 4;
    atomicAdd(dst + 0, e.x * vv);
    atomicAdd(dst + 1, e.y * vv);
    atomicAdd(dst + 2, e.z * vv);
    atomicAdd(dst + 3, e.w * vv);
}

// ---------------------------------------------------------------------------
// score = softmax(user_cur @ latent^T) : one wave per 16-user tile.
// Two v_wmma_f32_16x16x32_f16 cover K=64; only N=0..3 of C are used.
// ---------------------------------------------------------------------------
__global__ void score_kernel(const float* __restrict__ ucur,
                             const float* __restrict__ latent,
                             float* __restrict__ score) {
    __shared__ float sc[16][F_DIM];
    int lane = threadIdx.x & 31;
    int u0 = blockIdx.x * 16;          // 6250 tiles exactly
    int m = lane & 15, hi = lane >> 4;

    const float* ur = ucur + (u0 + m) * C_DIM;
    v16h a0, a1, b0, b1;
    #pragma unroll
    for (int h = 0; h < 16; ++h) {
        // A (16x32 f16): halves 0..7 -> K=8*hi+h ; halves 8..15 -> K=8*hi+h+8(+16)
        int k = hi * 8 + (h < 8 ? h : h + 8);
        a0[h] = (_Float16)ur[k];
        a1[h] = (_Float16)ur[32 + k];
        // B (32x16 f16): same K order as A, N = m; latent^T[k][n]=latent[n][k]
        int kb = hi * 8 + (h < 8 ? h : h + 8);
        float bv0 = (m < F_DIM) ? latent[m * C_DIM + kb]      : 0.0f;
        float bv1 = (m < F_DIM) ? latent[m * C_DIM + 32 + kb] : 0.0f;
        b0[h] = (_Float16)bv0;
        b1[h] = (_Float16)bv1;
    }
    v8f c = {};
    c = __builtin_amdgcn_wmma_f32_16x16x32_f16(false, a0, false, b0, (short)0, c, false, false);
    c = __builtin_amdgcn_wmma_f32_16x16x32_f16(false, a1, false, b1, (short)0, c, false, false);

    // C layout: VGPR v, lane -> row v + 8*hi, col m
    if (m < F_DIM) {
        #pragma unroll
        for (int v = 0; v < 8; ++v) sc[v + 8 * hi][m] = c[v];
    }
    __syncthreads();
    if (lane < 16) {
        float s0 = sc[lane][0], s1 = sc[lane][1], s2 = sc[lane][2], s3 = sc[lane][3];
        float mx = fmaxf(fmaxf(s0, s1), fmaxf(s2, s3));
        float e0 = expf(s0 - mx), e1 = expf(s1 - mx), e2 = expf(s2 - mx), e3 = expf(s3 - mx);
        float inv = 1.0f / (e0 + e1 + e2 + e3);
        float* dst = score + (u0 + lane) * F_DIM;
        dst[0] = e0 * inv; dst[1] = e1 * inv; dst[2] = e2 * inv; dst[3] = e3 * inv;
    }
}

// ---------------------------------------------------------------------------
// user update: P = score @ disen (v_wmma_f32_16x16x4_f32, exact f32, K=4),
// u = l2norm(agg_u * (1 + P)); user_cur = u; out_user += u.
// One wave per 16-user tile; 4 N-tiles of 16 cols cover C=64.
// ---------------------------------------------------------------------------
__global__ void user_update_kernel(const float* __restrict__ score,
                                   const float* __restrict__ disen,
                                   const float* __restrict__ aggu,
                                   float* __restrict__ ucur,
                                   float* __restrict__ outu) {
    int lane = threadIdx.x & 31;
    int u0 = blockIdx.x * 16;
    int m = lane & 15, hi = lane >> 4;

    // A (16x4 f32): VGPR0 = K=2*hi, VGPR1 = K=2*hi+1, rows m
    v2f a;
    a.x = score[(u0 + m) * F_DIM + 2 * hi + 0];
    a.y = score[(u0 + m) * F_DIM + 2 * hi + 1];

    float vals[4][8];
    float ss[8] = {0, 0, 0, 0, 0, 0, 0, 0};

    #pragma unroll
    for (int t = 0; t < 4; ++t) {
        int n = t * 16 + m;
        // B (4x16 f32): VGPR0 = K=2*hi row, VGPR1 = K=2*hi+1 row, col n
        v2f b;
        b.x = disen[(2 * hi + 0) * C_DIM + n];
        b.y = disen[(2 * hi + 1) * C_DIM + n];
        v8f c = {};
        c = __builtin_amdgcn_wmma_f32_16x16x4_f32(false, a, false, b, (short)0, c, false, false);
        #pragma unroll
        for (int v = 0; v < 8; ++v) {
            int row = v + 8 * hi;
            float g = aggu[(u0 + row) * C_DIM + n];
            float o = g * (1.0f + c[v]);   // agg*(score@disen) + agg
            vals[t][v] = o;
            ss[v] += o * o;
        }
    }
    // reduce sum-of-squares across each 16-lane half (rows are per-half)
    #pragma unroll
    for (int mask = 1; mask <= 8; mask <<= 1) {
        #pragma unroll
        for (int v = 0; v < 8; ++v) ss[v] += __shfl_xor(ss[v], mask);
    }
    float inv[8];
    #pragma unroll
    for (int v = 0; v < 8; ++v) inv[v] = 1.0f / fmaxf(sqrtf(ss[v]), EPS_F);
    #pragma unroll
    for (int t = 0; t < 4; ++t) {
        #pragma unroll
        for (int v = 0; v < 8; ++v) {
            int idx = (u0 + v + 8 * hi) * C_DIM + t * 16 + m;
            float o = vals[t][v] * inv[v];
            ucur[idx] = o;
            outu[idx] += o;
        }
    }
}

// ---------------------------------------------------------------------------
// entity update: e = l2norm(agg/cnt if cnt>0 else 0); entity_cur = e; out += e
// One wave per entity row (32 lanes x float2 = 64 cols).
// ---------------------------------------------------------------------------
__global__ void entity_update_kernel(const float* __restrict__ agg,
                                     const float* __restrict__ cnt,
                                     float* __restrict__ ecur,
                                     float* __restrict__ oute, int ne) {
    int wid = (blockIdx.x * blockDim.x + threadIdx.x) >> 5;
    int lane = threadIdx.x & 31;
    if (wid >= ne) return;
    float c = cnt[wid];
    int idx = wid * C_DIM + lane * 2;
    float2 g = *((const float2*)(agg + idx));
    float vx, vy;
    if (c > 0.0f) { float ic = 1.0f / c; vx = g.x * ic; vy = g.y * ic; }
    else          { vx = 0.0f; vy = 0.0f; }
    float ss = vx * vx + vy * vy;
    #pragma unroll
    for (int mask = 1; mask <= 16; mask <<= 1) ss += __shfl_xor(ss, mask);
    float inv = 1.0f / fmaxf(sqrtf(ss), EPS_F);
    float ox = vx * inv, oy = vy * inv;
    ecur[idx] = ox;  ecur[idx + 1] = oy;
    oute[idx] += ox; oute[idx + 1] += oy;
}

// ---------------------------------------------------------------------------
extern "C" void kernel_launch(void* const* d_in, const int* in_sizes, int n_in,
                              void* d_out, int out_size, void* d_ws, size_t ws_size,
                              hipStream_t stream) {
    const float* user_emb   = (const float*)d_in[0];
    const float* entity_emb = (const float*)d_in[1];
    const float* latent_emb = (const float*)d_in[2];
    const int*   edge_index = (const int*)  d_in[3];   // [2, E]
    const int*   edge_type  = (const int*)  d_in[4];
    const int*   ui_rows    = (const int*)  d_in[5];
    const int*   ui_cols    = (const int*)  d_in[6];
    const float* ui_vals    = (const float*)d_in[7];
    const float* weight     = (const float*)d_in[8];   // [9, 64]
    const float* disen_att  = (const float*)d_in[9];   // [4, 9]

    const int ENT_N  = N_ENT   * C_DIM;   // 12,800,000
    const int USER_N = N_USERS * C_DIM;   //  6,400,000

    float* out        = (float*)d_out;
    float* out_entity = out;
    float* out_user   = out + ENT_N;
    float* out_cor    = out + ENT_N + USER_N;

    float* ws = (float*)d_ws;
    float* entity_cur = ws;                     ws += ENT_N;
    float* user_cur   = ws;                     ws += USER_N;
    float* agg_e      = ws;                     ws += ENT_N;
    float* agg_u      = ws;                     ws += USER_N;
    float* cnt_e      = ws;                     ws += N_ENT;
    float* score_b    = ws;                     ws += N_USERS * F_DIM;
    float* disen_b    = ws;                     ws += F_DIM * C_DIM;

    const int* head = edge_index;            // edge_index[0]
    const int* tail = edge_index + E_EDGES;  // edge_index[1]

    // residual init + hop-invariant pieces
    init_copy_kernel<<<2048, 256, 0, stream>>>(entity_emb, entity_cur, out_entity, ENT_N);
    init_copy_kernel<<<2048, 256, 0, stream>>>(user_emb,   user_cur,   out_user,   USER_N);
    cor_kernel<<<1, 1, 0, stream>>>(disen_att, out_cor);
    disen_kernel<<<1, C_DIM, 0, stream>>>(disen_att, weight, disen_b);

    for (int hop = 0; hop < 2; ++hop) {
        hipMemsetAsync(agg_e, 0, (size_t)ENT_N  * sizeof(float), stream);
        hipMemsetAsync(agg_u, 0, (size_t)USER_N * sizeof(float), stream);
        hipMemsetAsync(cnt_e, 0, (size_t)N_ENT  * sizeof(float), stream);

        edge_agg_kernel<<<(E_EDGES * 16) / 256, 256, 0, stream>>>(
            head, tail, edge_type, entity_cur, weight, agg_e, cnt_e, E_EDGES);
        ui_agg_kernel<<<(NNZ_UI * 16) / 256, 256, 0, stream>>>(
            ui_rows, ui_cols, ui_vals, entity_cur, agg_u, NNZ_UI);

        score_kernel<<<N_USERS / 16, 32, 0, stream>>>(user_cur, latent_emb, score_b);
        user_update_kernel<<<N_USERS / 16, 32, 0, stream>>>(
            score_b, disen_b, agg_u, user_cur, out_user);
        entity_update_kernel<<<N_ENT / 8, 256, 0, stream>>>(
            agg_e, cnt_e, entity_cur, out_entity, N_ENT);
    }
}
